// SparseCausalAttentionBlock_8392366096621
// MI455X (gfx1250) — matrix operand order repaired and checked by reading, not compiled, not run on hardware
//
#include <hip/hip_runtime.h>
#include <hip/hip_bf16.h>

typedef __attribute__((ext_vector_type(16))) _Float16 v16h;
typedef __attribute__((ext_vector_type(8)))  _Float16 v8h;
typedef __attribute__((ext_vector_type(8)))  float    v8f;

__device__ __forceinline__ v8f wmma16(v16h a, v16h b, v8f c) {
    // D(16x16 f32) = A(16x32 f16) x B(32x16 f16) + C
    return __builtin_amdgcn_wmma_f32_16x16x32_f16(false, a, false, b, (short)0, c,
                                                  false, false);
}

// A-layout fragment (16x32 f16) from row-major f16; `p` -> (row, k0).
// elems 0..15 <-> k = half*16+0..15
__device__ __forceinline__ v16h load_a16(const _Float16* __restrict__ p, int half) {
    v8h lo = *(const v8h*)(p + half * 16);
    v8h hi = *(const v8h*)(p + half * 16 + 8);
    return __builtin_shufflevector(lo, hi, 0, 1, 2, 3, 4, 5, 6, 7,
                                   8, 9, 10, 11, 12, 13, 14, 15);
}

// B-layout fragment (32x16 f16): lane = column n; per-lane k = half*16 + e.
// For C = A*W^T with W row-major [N,K]: one aligned 32B load from row n of W.
__device__ __forceinline__ v16h load_b16(const _Float16* __restrict__ p, int half) {
    return *(const v16h*)(p + half * 16);
}

// ---------------------------------------------------------------------------
// elementwise f32 -> f16 (n multiple of 8)
// ---------------------------------------------------------------------------
__global__ __launch_bounds__(256) void cvt_f32_f16(const float* __restrict__ src,
                                                   _Float16* __restrict__ dst,
                                                   long n) {
    const long i = ((long)blockIdx.x * 256 + threadIdx.x) * 8;
    if (i >= n) return;
    float4 a = *(const float4*)(src + i);
    float4 b = *(const float4*)(src + i + 4);
    v8h o;
    o[0] = (_Float16)a.x; o[1] = (_Float16)a.y;
    o[2] = (_Float16)a.z; o[3] = (_Float16)a.w;
    o[4] = (_Float16)b.x; o[5] = (_Float16)b.y;
    o[6] = (_Float16)b.z; o[7] = (_Float16)b.w;
    *(v8h*)(dst + i) = o;
}

// copy f32 -> f32 and f16 in one pass (n multiple of 8)
__global__ __launch_bounds__(256) void copy_cvt(const float* __restrict__ src,
                                                float* __restrict__ dstf,
                                                _Float16* __restrict__ dsth,
                                                long n) {
    const long i = ((long)blockIdx.x * 256 + threadIdx.x) * 8;
    if (i >= n) return;
    float4 a = *(const float4*)(src + i);
    float4 b = *(const float4*)(src + i + 4);
    *(float4*)(dstf + i)     = a;
    *(float4*)(dstf + i + 4) = b;
    v8h o;
    o[0] = (_Float16)a.x; o[1] = (_Float16)a.y;
    o[2] = (_Float16)a.z; o[3] = (_Float16)a.w;
    o[4] = (_Float16)b.x; o[5] = (_Float16)b.y;
    o[6] = (_Float16)b.z; o[7] = (_Float16)b.w;
    *(v8h*)(dsth + i) = o;
}

// ---------------------------------------------------------------------------
// GEMM: C[M,N] = act(A[M,K] @ W[N,K]^T + bias) (+ R)   (A, W in f16)
// one wave -> 32x64 output tile (8 WMMAs / 32-k-step); 8 waves / block.
// Waves of a block share the same B panel (consecutive tm, same tn).
// OUTH: store C as f16, else f32. OUT2H: additionally store f16 copy to C2.
// ---------------------------------------------------------------------------
template <bool BIAS, bool GELU_ACT, bool RESID, bool OUTH, bool OUT2H>
__global__ __launch_bounds__(256) void gemm_f16wmma(
    const _Float16* __restrict__ A, const _Float16* __restrict__ W,
    const float* __restrict__ bias, const float* __restrict__ R,
    void* __restrict__ Cv, _Float16* __restrict__ C2, int M, int N, int K) {
    const int lane = threadIdx.x & 31;
    const int wave = threadIdx.x >> 5;
    const int ntm  = M >> 5;
    const long tile = (long)blockIdx.x * 8 + wave;
    if (tile >= (long)ntm * (N >> 6)) return;
    const int tn   = (int)(tile / ntm);  // same tn across the 8 waves of a block
    const int tm   = (int)(tile % ntm);
    const int cn   = lane & 15;
    const int half = lane >> 4;

    v8f a00 = {}, a01 = {}, a02 = {}, a03 = {};
    v8f a10 = {}, a11 = {}, a12 = {}, a13 = {};
    const _Float16* ar0 = A + (long)(tm * 32 + cn) * K;
    const _Float16* ar1 = ar0 + (long)16 * K;
    const _Float16* wr0 = W + (long)(tn * 64 + cn) * K;
    const _Float16* wr1 = wr0 + (long)16 * K;
    const _Float16* wr2 = wr0 + (long)32 * K;
    const _Float16* wr3 = wr0 + (long)48 * K;

#pragma unroll 2
    for (int k = 0; k < K; k += 32) {
        v16h f0 = load_a16(ar0 + k, half);
        v16h f1 = load_a16(ar1 + k, half);
        v16h b0 = load_b16(wr0 + k, half);
        v16h b1 = load_b16(wr1 + k, half);
        v16h b2 = load_b16(wr2 + k, half);
        v16h b3 = load_b16(wr3 + k, half);
        a00 = wmma16(f0, b0, a00);
        a10 = wmma16(f1, b0, a10);
        a01 = wmma16(f0, b1, a01);
        a11 = wmma16(f1, b1, a11);
        a02 = wmma16(f0, b2, a02);
        a12 = wmma16(f1, b2, a12);
        a03 = wmma16(f0, b3, a03);
        a13 = wmma16(f1, b3, a13);
    }

    v8f accs[2][4] = {{a00, a01, a02, a03}, {a10, a11, a12, a13}};
    float* Cf = (float*)Cv;
    _Float16* Ch = (_Float16*)Cv;
#pragma unroll
    for (int g = 0; g < 2; ++g) {
#pragma unroll
        for (int f = 0; f < 4; ++f) {
            const int col = tn * 64 + f * 16 + cn;
            const float bv = BIAS ? bias[col] : 0.0f;
#pragma unroll
            for (int r = 0; r < 8; ++r) {
                const long orow = (long)(tm * 32 + g * 16 + r + half * 8);
                float v = accs[g][f][r] + bv;
                if (GELU_ACT) v = 0.5f * v * (1.0f + erff(v * 0.70710678118654752f));
                if (RESID) v += R[orow * N + col];
                if (OUTH) Ch[orow * N + col] = (_Float16)v;
                else      Cf[orow * N + col] = v;
                if (OUT2H) C2[orow * N + col] = (_Float16)v;
            }
        }
    }
}

// ---------------------------------------------------------------------------
// Local windowed-causal attention. QKV: [64*512, 1536] f16 (Q|K|V packed).
// one wave per (b, h, 16-query tile); fixed 6 key-tiles (96 keys) band.
// ---------------------------------------------------------------------------
__global__ __launch_bounds__(32) void local_attn(const _Float16* __restrict__ QKV,
                                                 _Float16* __restrict__ O) {
    __shared__ __align__(16) _Float16 P[16 * 96];
    const int lane = threadIdx.x;
    const int cn   = lane & 15;
    const int half = lane >> 4;
    const int tq   = blockIdx.x & 31;
    const int h    = (blockIdx.x >> 5) & 7;
    const int b    = blockIdx.x >> 8;
    const int mBase = tq * 16;
    const long tok  = (long)b * 512;
    const int LD    = 1536;

    v16h qa0, qa1;
    {
        const _Float16* qp = QKV + (tok + mBase + cn) * LD + h * 64;
        qa0 = load_a16(qp, half);
        qa1 = load_a16(qp + 32, half);
    }

    float sc[6][8];
#pragma unroll
    for (int t = 0; t < 6; ++t) {
        const int tt = tq - 5 + t;
        if (tt < 0) {
#pragma unroll
            for (int r = 0; r < 8; ++r) sc[t][r] = -1e30f;
            continue;
        }
        const _Float16* kp = QKV + (tok + tt * 16 + cn) * LD + 512 + h * 64;
        v8f acc = {};
        acc = wmma16(qa0, load_b16(kp, half), acc);
        acc = wmma16(qa1, load_b16(kp + 32, half), acc);
#pragma unroll
        for (int r = 0; r < 8; ++r) {
            const int i = mBase + r + half * 8;
            const int j = tt * 16 + cn;
            // allowed iff j <= i && j >= i-64  <=>  (unsigned)(i-j) <= 64
            sc[t][r] = ((unsigned)(i - j) <= 64u) ? acc[r] * 0.125f : -1e30f;
        }
    }

    // softmax across 6 tiles x 16 lanes (xor 1,2,4,8 stays within 16-lane group)
#pragma unroll
    for (int r = 0; r < 8; ++r) {
        float mx = -1e30f;
#pragma unroll
        for (int t = 0; t < 6; ++t) mx = fmaxf(mx, sc[t][r]);
#pragma unroll
        for (int m = 1; m <= 8; m <<= 1) mx = fmaxf(mx, __shfl_xor(mx, m, 32));
        float pv[6];
        float sum = 0.0f;
#pragma unroll
        for (int t = 0; t < 6; ++t) {
            pv[t] = __expf(sc[t][r] - mx);
            sum += pv[t];
        }
#pragma unroll
        for (int m = 1; m <= 8; m <<= 1) sum += __shfl_xor(sum, m, 32);
        const float inv = 1.0f / sum;
        const int prow  = r + half * 8;
#pragma unroll
        for (int t = 0; t < 6; ++t)
            P[prow * 96 + t * 16 + cn] = (_Float16)(pv[t] * inv);
    }
    __syncthreads();

    const int kb0 = mBase - 80;  // first key of band (may be <0; p==0 there)
#pragma unroll
    for (int f = 0; f < 4; ++f) {
        v8f acc = {};
#pragma unroll
        for (int c = 0; c < 3; ++c) {
            v16h pa = load_a16(&P[cn * 96 + c * 32], half);
            v16h vb;
#pragma unroll
            for (int e = 0; e < 16; ++e) {
                int key = kb0 + c * 32 + half * 16 + e;
                key = key < 0 ? 0 : key;  // p==0 for these, value irrelevant
                vb[e] = QKV[(tok + key) * LD + 1024 + h * 64 + f * 16 + cn];
            }
            acc = wmma16(pa, vb, acc);
        }
#pragma unroll
        for (int r = 0; r < 8; ++r) {
            const int i = mBase + r + half * 8;
            O[(tok + i) * 512 + h * 64 + f * 16 + cn] = (_Float16)acc[r];
        }
    }
}

// ---------------------------------------------------------------------------
// Global attention to 32 anchors. Q: [64*512,512] f16, KV: [64*32,1024] f16.
// ---------------------------------------------------------------------------
__global__ __launch_bounds__(32) void global_attn(const _Float16* __restrict__ Q,
                                                  const _Float16* __restrict__ KV,
                                                  _Float16* __restrict__ O) {
    __shared__ __align__(16) _Float16 P[16 * 32];
    const int lane = threadIdx.x;
    const int cn   = lane & 15;
    const int half = lane >> 4;
    const int tq   = blockIdx.x & 31;
    const int h    = (blockIdx.x >> 5) & 7;
    const int b    = blockIdx.x >> 8;
    const int mBase = tq * 16;
    const long tok  = (long)b * 512;
    const long atok = (long)b * 32;

    v16h qa0, qa1;
    {
        const _Float16* qp = Q + (tok + mBase + cn) * 512 + h * 64;
        qa0 = load_a16(qp, half);
        qa1 = load_a16(qp + 32, half);
    }

    float sc[2][8];
#pragma unroll
    for (int t = 0; t < 2; ++t) {
        const _Float16* kp = KV + (atok + t * 16 + cn) * 1024 + h * 64;
        v8f acc = {};
        acc = wmma16(qa0, load_b16(kp, half), acc);
        acc = wmma16(qa1, load_b16(kp + 32, half), acc);
#pragma unroll
        for (int r = 0; r < 8; ++r) sc[t][r] = acc[r] * 0.125f;
    }

#pragma unroll
    for (int r = 0; r < 8; ++r) {
        float mx = fmaxf(sc[0][r], sc[1][r]);
#pragma unroll
        for (int m = 1; m <= 8; m <<= 1) mx = fmaxf(mx, __shfl_xor(mx, m, 32));
        float e0 = __expf(sc[0][r] - mx);
        float e1 = __expf(sc[1][r] - mx);
        float sum = e0 + e1;
#pragma unroll
        for (int m = 1; m <= 8; m <<= 1) sum += __shfl_xor(sum, m, 32);
        const float inv = 1.0f / sum;
        const int prow  = r + half * 8;
        P[prow * 32 + cn]      = (_Float16)(e0 * inv);
        P[prow * 32 + 16 + cn] = (_Float16)(e1 * inv);
    }
    __syncthreads();

#pragma unroll
    for (int f = 0; f < 4; ++f) {
        v16h pa = load_a16(&P[cn * 32], half);
        v16h vb;
#pragma unroll
        for (int e = 0; e < 16; ++e) {
            const int key = half * 16 + e;
            vb[e] = KV[(atok + key) * 1024 + 512 + h * 64 + f * 16 + cn];
        }
        v8f acc = {};
        acc = wmma16(pa, vb, acc);
#pragma unroll
        for (int r = 0; r < 8; ++r) {
            const int i = mBase + r + half * 8;
            O[(tok + i) * 512 + h * 64 + f * 16 + cn] = (_Float16)acc[r];
        }
    }
}

// ---------------------------------------------------------------------------
// In-place LayerNorm over 512-wide rows; one wave per row.
// DUAL: also emit an f16 copy of the normalized row.
// ---------------------------------------------------------------------------
template <bool DUAL>
__global__ __launch_bounds__(256) void layernorm512(float* __restrict__ X,
                                                    const float* __restrict__ g,
                                                    const float* __restrict__ be,
                                                    _Float16* __restrict__ Xh,
                                                    long rows) {
    const long rid = (long)blockIdx.x * 8 + (threadIdx.x >> 5);
    if (rid >= rows) return;
    const int lane = threadIdx.x & 31;
    float* xr = X + rid * 512;
    float v[16];
    float s = 0.0f, s2 = 0.0f;
#pragma unroll
    for (int i = 0; i < 16; ++i) {
        v[i] = xr[i * 32 + lane];
        s += v[i];
        s2 += v[i] * v[i];
    }
#pragma unroll
    for (int m = 1; m <= 16; m <<= 1) {
        s  += __shfl_xor(s, m, 32);
        s2 += __shfl_xor(s2, m, 32);
    }
    const float mu  = s * (1.0f / 512.0f);
    const float var = s2 * (1.0f / 512.0f) - mu * mu;
    const float inv = rsqrtf(var + 1e-5f);
#pragma unroll
    for (int i = 0; i < 16; ++i) {
        const int c = i * 32 + lane;
        const float y = (v[i] - mu) * inv * g[c] + be[c];
        xr[c] = y;
        if (DUAL) Xh[rid * 512 + c] = (_Float16)y;
    }
}

extern "C" void kernel_launch(void* const* d_in, const int* in_sizes, int n_in,
                              void* d_out, int out_size, void* d_ws, size_t ws_size,
                              hipStream_t stream) {
    const float* x       = (const float*)d_in[0];
    const float* anchors = (const float*)d_in[1];
    const float* lw_in   = (const float*)d_in[2];
    const float* lb_in   = (const float*)d_in[3];
    const float* lw_out  = (const float*)d_in[4];
    const float* lb_out  = (const float*)d_in[5];
    const float* gw_in   = (const float*)d_in[6];
    const float* gb_in   = (const float*)d_in[7];
    const float* gw_out  = (const float*)d_in[8];
    const float* gb_out  = (const float*)d_in[9];
    const float* w1      = (const float*)d_in[10];
    const float* b1      = (const float*)d_in[11];
    const float* w2      = (const float*)d_in[12];
    const float* b2      = (const float*)d_in[13];
    const float* g1      = (const float*)d_in[14];
    const float* be1     = (const float*)d_in[15];
    const float* g2      = (const float*)d_in[16];
    const float* be2     = (const float*)d_in[17];
    float* out = (float*)d_out;

    const long T  = 64L * 512;  // 32768 tokens
    const long TA = 64L * 32;   // 2048 anchor tokens

    // ---- workspace layout ----
    float*    X    = (float*)d_ws;             // [T,512] f32 residual (64 MB)
    _Float16* Xh   = (_Float16*)(X + T * 512); // [T,512]      (32 MB)
    _Float16* QKVh = Xh + T * 512;             // [T,1536]     (96 MB)
    _Float16* ATTh = QKVh + T * 1536;          // [T,512]      (32 MB)
    _Float16* ANCh = ATTh + T * 512;           // [TA,512]     (2 MB)
    _Float16* lw_in_h  = ANCh + TA * 512;      // converted weights (8 MB total)
    _Float16* lw_out_h = lw_in_h + 1536L * 512;
    _Float16* gw_in_h  = lw_out_h + 512L * 512;
    _Float16* gw_out_h = gw_in_h + 1536L * 512;
    _Float16* w1_h     = gw_out_h + 512L * 512;
    _Float16* w2_h     = w1_h + 2048L * 512;
    _Float16* Hh   = QKVh;                     // [T,2048] overlays QKV+ATT
    _Float16* Qgh  = QKVh;                     // [T,512]
    _Float16* KVgh = QKVh + T * 512;           // [TA,1024]

    auto cblk = [](long n) { return (int)((n / 8 + 255) / 256); };
    auto nblk = [](long M, long N) { return (int)(((M / 32) * (N / 64)) / 8); };

    // ---- init: residual stream (f32 + f16) and one-time f16 staging ----
    copy_cvt<<<cblk(T * 512), 256, 0, stream>>>(x, X, Xh, T * 512);
    cvt_f32_f16<<<cblk(1536L * 512), 256, 0, stream>>>(lw_in,  lw_in_h,  1536L * 512);
    cvt_f32_f16<<<cblk(512L * 512),  256, 0, stream>>>(lw_out, lw_out_h, 512L * 512);
    cvt_f32_f16<<<cblk(1536L * 512), 256, 0, stream>>>(gw_in,  gw_in_h,  1536L * 512);
    cvt_f32_f16<<<cblk(512L * 512),  256, 0, stream>>>(gw_out, gw_out_h, 512L * 512);
    cvt_f32_f16<<<cblk(2048L * 512), 256, 0, stream>>>(w1, w1_h, 2048L * 512);
    cvt_f32_f16<<<cblk(512L * 2048), 256, 0, stream>>>(w2, w2_h, 512L * 2048);
    cvt_f32_f16<<<cblk(TA * 512),    256, 0, stream>>>(anchors, ANCh, TA * 512);

    // 1) local QKV projection (f16 out)
    gemm_f16wmma<true, false, false, true, false><<<nblk(T, 1536), 256, 0, stream>>>(
        Xh, lw_in_h, lb_in, nullptr, QKVh, nullptr, (int)T, 1536, 512);
    // 2) local windowed-causal attention
    local_attn<<<64 * 8 * 32, 32, 0, stream>>>(QKVh, ATTh);
    // 3) local out-proj + residual -> X (f32) and Xh (f16)
    gemm_f16wmma<true, false, true, false, true><<<nblk(T, 512), 256, 0, stream>>>(
        ATTh, lw_out_h, lb_out, X, X, Xh, (int)T, 512, 512);
    // 4) global Q projection (f16 out)
    gemm_f16wmma<true, false, false, true, false><<<nblk(T, 512), 256, 0, stream>>>(
        Xh, gw_in_h, gb_in, nullptr, Qgh, nullptr, (int)T, 512, 512);
    // 5) anchor K,V projection (rows 512..1535 of packed gw_in)
    gemm_f16wmma<true, false, false, true, false><<<nblk(TA, 1024), 256, 0, stream>>>(
        ANCh, gw_in_h + 512L * 512, gb_in + 512, nullptr, KVgh, nullptr,
        (int)TA, 1024, 512);
    // 6) global attention
    global_attn<<<64 * 8 * 32, 32, 0, stream>>>(Qgh, KVgh, ATTh);
    // 7) global out-proj + residual -> X
    gemm_f16wmma<true, false, true, false, false><<<nblk(T, 512), 256, 0, stream>>>(
        ATTh, gw_out_h, gb_out, X, X, nullptr, (int)T, 512, 512);
    // 8) LayerNorm 1 (in place, dual f32+f16 output)
    layernorm512<true><<<(int)(T / 8), 256, 0, stream>>>(X, g1, be1, Xh, T);
    // 9) FFN up + exact GELU -> hidden [T,2048] f16
    gemm_f16wmma<true, true, false, true, false><<<nblk(T, 2048), 256, 0, stream>>>(
        Xh, w1_h, b1, nullptr, Hh, nullptr, (int)T, 2048, 512);
    // 10) FFN down + residual -> d_out (f32)
    gemm_f16wmma<true, false, true, false, false><<<nblk(T, 512), 256, 0, stream>>>(
        Hh, w2_h, b2, X, out, nullptr, (int)T, 512, 2048);
    // 11) LayerNorm 2 (in place on d_out)
    layernorm512<false><<<(int)(T / 8), 256, 0, stream>>>(out, g2, be2, nullptr, T);
}
